// HSMNet_41351945125913
// MI455X (gfx1250) — compile-verified, matrix operand order checked by reading, not run on hardware
//
#include <hip/hip_runtime.h>

// HSMNet feature-vol + channel reduce + softmax + disparity regression.
// B=2, C=32, H=96, W=160, maxdisp=384 -> D=48 (fixed by setup_inputs()).
//
// Strategy (MI455X): memory-bound problem (~8 MB total traffic, ~0.34 us at
// 23.3 TB/s). One workgroup per (b,h) row: load ref/tgt rows once to LDS as
// f16 [w][c], then reuse 48x. Channel reduction Sum_c |diff| done with
// v_wmma_f32_16x16x32_f16 against an all-ones B matrix (D[m][n] = Sum_k A[m,k]);
// since B is all ones, only the row mapping of the A layout matters, making
// this robust. f16 abs is a bit-mask (x & 0x7FFF) -> one v_and_b32 per packed
// dword instead of xor+cndmask chains. Softmax + expectation over D=48 stays
// in LDS/registers.

typedef __attribute__((ext_vector_type(16))) _Float16       v16h;
typedef __attribute__((ext_vector_type(8)))  _Float16       v8h;
typedef __attribute__((ext_vector_type(8)))  unsigned short v8u;
typedef __attribute__((ext_vector_type(8)))  float          v8f;

#define BB 2
#define CC 32
#define HH 96
#define WW 160
#define DD 48
#define CPAD 40          // halves per LDS row (80 B): keeps 16B chunks aligned
#define NTHREADS 256     // 8 wave32's

__device__ __forceinline__ v8h absdiff_h8(v8h x, v8h y) {
  v8h d = x - y;                       // v_pk_add_f16 (neg) x4
  v8u u = (v8u)d & (unsigned short)0x7FFF;  // v_and_b32 x4 (clear sign bits)
  return (v8h)u;
}

__global__ __launch_bounds__(NTHREADS) void hsmnet_disp_kernel(
    const float* __restrict__ ref, const float* __restrict__ tgt,
    float* __restrict__ out)
{
  __shared__ _Float16 refh[WW * CPAD];   // 12800 B
  __shared__ _Float16 tgth[WW * CPAD];   // 12800 B
  __shared__ float    score[DD * WW];    // 30720 B  (total 56320 B)

  const int t  = threadIdx.x;
  const int bh = blockIdx.x;
  const int b  = bh / HH;
  const int h  = bh - b * HH;
  const long rowBase = (long)b * (CC * HH * WW) + (long)h * WW;

  // ---- Stage 1: global f32 [c][w] row -> LDS f16 [w][c] (transposed) ----
  for (int idx = t; idx < CC * WW; idx += NTHREADS) {
    int c = idx / WW;
    int w = idx - c * WW;
    long g = rowBase + (long)c * (HH * WW) + w;
    __builtin_prefetch(&ref[g + 4 * HH * WW], 0, 0);   // global_prefetch_b8
    refh[w * CPAD + c] = (_Float16)ref[g];
    tgth[w * CPAD + c] = (_Float16)tgt[g];
  }
  __syncthreads();

  // ---- Stage 2: score[d][w] = -Sum_c |ref - tgt_shift| via WMMA ----
  const int wv    = t >> 5;
  const int lane  = t & 31;
  const int m     = lane & 15;   // A-matrix row handled by this lane
  const int hsel  = lane >> 4;   // which K/M half this lane covers
  const int kbase = hsel * 8;

  v16h bones;
  for (int e = 0; e < 16; ++e) bones[e] = (_Float16)1.0f;

  // 48 disparities x 10 tiles of 16 pixels = 480 tasks; 60 per wave (uniform,
  // so EXEC is all-ones at every WMMA as the ISA requires).
  for (int task = wv; task < DD * (WW / 16); task += (NTHREADS / 32)) {
    int d    = task / (WW / 16);
    int tile = task - d * (WW / 16);
    int w    = tile * 16 + m;
    int ws   = w - d;
    // w < d: cost must be exactly 0 -> read ref as "target" so diff == 0.
    const _Float16* tp = (ws >= 0) ? tgth : refh;
    int wsrc = (ws >= 0) ? ws : w;

    const v8h r0 = *(const v8h*)&refh[w * CPAD + kbase];        // K kbase..+7
    const v8h r1 = *(const v8h*)&refh[w * CPAD + 16 + kbase];   // K 16+kbase..
    const v8h t0 = *(const v8h*)&tp[wsrc * CPAD + kbase];
    const v8h t1 = *(const v8h*)&tp[wsrc * CPAD + 16 + kbase];

    v8h a0 = absdiff_h8(r0, t0);
    v8h a1 = absdiff_h8(r1, t1);
    v16h a = __builtin_shufflevector(a0, a1, 0, 1, 2, 3, 4, 5, 6, 7,
                                             8, 9, 10, 11, 12, 13, 14, 15);

    v8f c = {};
    // D[m][n] = Sum_k A[m,k] * 1  -> per-pixel channel sum, replicated over n
    c = __builtin_amdgcn_wmma_f32_16x16x32_f16(
        /*neg_a=*/false, a, /*neg_b=*/false, bones,
        /*c_mod=*/(short)0, c, /*reuse_a=*/false, /*reuse_b=*/false);

    if (m == 0) {  // lanes 0 (rows 0-7) and 16 (rows 8-15) hold column n=0
      int wbase = tile * 16 + hsel * 8;
      for (int v = 0; v < 8; ++v)
        score[d * WW + wbase + v] = -c[v];
    }
  }
  __syncthreads();

  // ---- Stage 3: softmax over d + disparity regression (stride 8) ----
  if (t < WW) {
    float mx = score[t];
    for (int d = 1; d < DD; ++d) {
      float s = score[d * WW + t];
      mx = (s > mx) ? s : mx;
    }
    float sum = 0.f, wsum = 0.f;
    for (int d = 0; d < DD; ++d) {
      float e = __expf(score[d * WW + t] - mx);
      sum  += e;
      wsum += e * (float)(d * 8);
    }
    out[(b * HH + h) * WW + t] = wsum / sum;
  }
}

extern "C" void kernel_launch(void* const* d_in, const int* in_sizes, int n_in,
                              void* d_out, int out_size, void* d_ws, size_t ws_size,
                              hipStream_t stream) {
  (void)in_sizes; (void)n_in; (void)d_ws; (void)ws_size; (void)out_size;
  const float* ref = (const float*)d_in[0];   // left_fea  [2,32,96,160] f32
  const float* tgt = (const float*)d_in[1];   // right_fea [2,32,96,160] f32
  // d_in[2] = maxdisp (384) -> D = 48, baked into the kernel.
  float* out = (float*)d_out;                 // pred [2,96,160] f32
  hsmnet_disp_kernel<<<BB * HH, NTHREADS, 0, stream>>>(ref, tgt, out);
}
